// LSTM_8650064134553
// MI455X (gfx1250) — matrix operand order repaired and checked by reading, not compiled, not run on hardware
//
#include <hip/hip_runtime.h>

// ---------------------------------------------------------------------------
// 2-layer LSTM + FC feedback, 32 steps, B=32768, D=128, H=64.
// MI455X / gfx1250 strategy:
//   * 256 workgroups x 128 batch rows; 8 wave32 per WG, 16 rows per wave.
//   * bf16 weights resident in LDS with PADDED row strides (136 / 72 els)
//     so every ds_load_b128 fragment load is bank-conflict-free.
//   * v_wmma_f32_16x16x32_bf16 everywhere; accumulators start from the
//     inline SRC2=0 form; biases folded into the activation stage.
//   * A fragments hoisted per layer (K-tile only); B fragments software-
//     pipelined one K-tile ahead so WMMAs never wait on dscnt==0.
//   * Cell state kept in WMMA accumulator register layout across all steps;
//     waves fully independent after init (no barriers in the time loop).
// ---------------------------------------------------------------------------

typedef __bf16 bf16_t;
typedef __attribute__((ext_vector_type(16))) __bf16 v16bf;
typedef __attribute__((ext_vector_type(8)))  __bf16 v8bf;
typedef __attribute__((ext_vector_type(8)))  float   v8f;

#define LSTM_D   128
#define LSTM_H   64
#define LSTM_G   256      // 4*H
#define BT       128      // batch rows per workgroup
#define NTHREADS 256      // 8 wave32

// Padded LDS row strides (elements): 136*2B = 272B = 68 DW, 72*2B = 144B = 36 DW
// -> 16 consecutive rows map to 16 disjoint 4-bank groups (conflict-free b128).
#define PD 136            // padded stride for K=128 buffers
#define PH 72             // padded stride for K=64  buffers

// LDS element offsets (bf16 elements)
static constexpr int SM_W0   = 0;                       // Wih0 [256][PD]
static constexpr int SM_WH0  = SM_W0  + LSTM_G * PD;    // Whh0 [256][PH]
static constexpr int SM_W1   = SM_WH0 + LSTM_G * PH;    // Wih1 [256][PH]
static constexpr int SM_WH1  = SM_W1  + LSTM_G * PH;    // Whh1 [256][PH]
static constexpr int SM_WFC  = SM_WH1 + LSTM_G * PH;    // Wfc  [128][PH]
static constexpr int SM_BF16_WEND = SM_WFC + LSTM_D * PH;   // 99328 els
static constexpr int SM_BIAS_BYTES = (256 + 256 + 128) * 4; // 2560 B
static constexpr int SM_STATE_ELS  = BT * PD + 4 * BT * PH; // y + 2x(h0,h1)
static constexpr size_t SMEM_BYTES =
    (size_t)SM_BF16_WEND * 2 + SM_BIAS_BYTES + (size_t)SM_STATE_ELS * 2; // 309760

__device__ __forceinline__ float fast_sigmoid(float x) {
    float e = __builtin_amdgcn_exp2f(-1.4426950408889634f * x);
    return __builtin_amdgcn_rcpf(1.0f + e);
}
__device__ __forceinline__ float fast_tanh(float x) {
    float e = __builtin_amdgcn_exp2f(-2.8853900817779268f * x); // exp(-2x)
    return 1.0f - 2.0f * e * __builtin_amdgcn_rcpf(1.0f + e);
}

__device__ __forceinline__ v16bf cat8(v8bf lo, v8bf hi) {
    return __builtin_shufflevector(lo, hi, 0,1,2,3,4,5,6,7,8,9,10,11,12,13,14,15);
}

// A fragment (16x32 bf16, M rows = batch): lane L (L%16 = row m) holds
// K = [h*8 .. h*8+7] and [h*8+16 .. h*8+23], h = L/16.  Row-major padded LDS.
__device__ __forceinline__ v16bf load_a_frag(const bf16_t* row_base, int stride,
                                             int m, int half, int k0) {
    const bf16_t* p = row_base + m * stride + k0 + half * 8;
    v8bf lo = *(const v8bf*)(p);
    v8bf hi = *(const v8bf*)(p + 16);
    return cat8(lo, hi);
}

// B fragment (32x16 bf16, N cols = gate units): lane L (L%16 = col n) holds
// K = [h*8 .. h*8+7] and [h*8+16 .. h*8+23].  Weights stored [n][k] row-major, padded stride.
__device__ __forceinline__ v16bf load_b_frag(const bf16_t* w, int stride,
                                             int n, int half, int k0) {
    const bf16_t* p = w + n * stride + k0 + half * 8;
    v8bf lo = *(const v8bf*)(p);
    v8bf hi = *(const v8bf*)(p + 16);
    return cat8(lo, hi);
}

#define WMMA_BF16(a, b, c) \
    __builtin_amdgcn_wmma_f32_16x16x32_bf16(false, (a), false, (b), (short)0, (c), false, false)

__global__ __launch_bounds__(NTHREADS)
void lstm2_wmma_kernel(const float* __restrict__ x,
                       const float* __restrict__ Wih0,
                       const float* __restrict__ Whh0,
                       const float* __restrict__ b0,
                       const float* __restrict__ Wih1,
                       const float* __restrict__ Whh1,
                       const float* __restrict__ b1,
                       const float* __restrict__ Wfc,
                       const float* __restrict__ bfc,
                       const int*   __restrict__ np,
                       float* __restrict__ out,
                       int Btot)
{
    extern __shared__ char smem_raw[];
    bf16_t* sw    = (bf16_t*)smem_raw;
    bf16_t* sWih0 = sw + SM_W0;
    bf16_t* sWhh0 = sw + SM_WH0;
    bf16_t* sWih1 = sw + SM_W1;
    bf16_t* sWhh1 = sw + SM_WH1;
    bf16_t* sWfc  = sw + SM_WFC;
    float*  sb0   = (float*)(smem_raw + (size_t)SM_BF16_WEND * 2);
    float*  sb1   = sb0 + 256;
    float*  sbfc  = sb0 + 512;
    bf16_t* sY    = (bf16_t*)(smem_raw + (size_t)SM_BF16_WEND * 2 + SM_BIAS_BYTES);
    bf16_t* sH0a  = sY   + BT * PD;
    bf16_t* sH0b  = sH0a + BT * PH;
    bf16_t* sH1a  = sH0b + BT * PH;
    bf16_t* sH1b  = sH1a + BT * PH;

    const int tid = threadIdx.x;
    const size_t row0 = (size_t)blockIdx.x * BT;

    // ---- one-time: weights f32 -> bf16 into padded LDS; x -> y; zero h ----
    for (int i = tid; i < LSTM_G * LSTM_D; i += NTHREADS)
        sWih0[(i >> 7) * PD + (i & 127)] = (bf16_t)Wih0[i];
    for (int i = tid; i < LSTM_G * LSTM_H; i += NTHREADS) {
        const int r = i >> 6, c = i & 63;
        sWhh0[r * PH + c] = (bf16_t)Whh0[i];
        sWih1[r * PH + c] = (bf16_t)Wih1[i];
        sWhh1[r * PH + c] = (bf16_t)Whh1[i];
    }
    for (int i = tid; i < LSTM_D * LSTM_H; i += NTHREADS)
        sWfc[(i >> 6) * PH + (i & 63)] = (bf16_t)Wfc[i];
    for (int i = tid; i < LSTM_G; i += NTHREADS) { sb0[i] = b0[i]; sb1[i] = b1[i]; }
    for (int i = tid; i < LSTM_D; i += NTHREADS) sbfc[i] = bfc[i];
    for (int i = tid; i < BT * LSTM_D; i += NTHREADS)
        sY[(i >> 7) * PD + (i & 127)] = (bf16_t)x[row0 * LSTM_D + i];
    for (int i = tid; i < BT * PH; i += NTHREADS) {
        sH0a[i] = (bf16_t)0.f; sH1a[i] = (bf16_t)0.f;
    }
    __syncthreads();

    const int wave = tid >> 5;
    const int lane = tid & 31;
    const int half = lane >> 4;   // K / row split within the wave
    const int ln   = lane & 15;   // row (A) or column (B/D) within tile
    const int wrow = wave * 16;   // this wave's local row tile

    bf16_t* h0_rd = sH0a; bf16_t* h0_wr = sH0b;
    bf16_t* h1_rd = sH1a; bf16_t* h1_wr = sH1b;

    // Loop-invariant per-lane bias values: hoisted out of the time loop.
    float bi0[4], bf0[4], bg0[4], bo0[4];
    float bi1[4], bf1[4], bg1[4], bo1[4];
    float byc[8];
    #pragma unroll
    for (int nh = 0; nh < 4; ++nh) {
        const int ni = nh * 16 + ln;
        bi0[nh] = sb0[ni];  bf0[nh] = sb0[64 + ni];
        bg0[nh] = sb0[128 + ni];  bo0[nh] = sb0[192 + ni];
        bi1[nh] = sb1[ni];  bf1[nh] = sb1[64 + ni];
        bg1[nh] = sb1[128 + ni];  bo1[nh] = sb1[192 + ni];
    }
    #pragma unroll
    for (int ny = 0; ny < 8; ++ny) byc[ny] = sbfc[ny * 16 + ln];

    // cell state in WMMA accumulator layout, resident across all timesteps
    v8f c0[4], c1[4];
    #pragma unroll
    for (int j = 0; j < 4; ++j) { c0[j] = {}; c1[j] = {}; }

    const int n = *np;
    const size_t plane = (size_t)Btot * LSTM_D;
    float* y_stack = out + plane;   // [n][B][D]
    float* y_last  = out;           // [B][D]

    const bf16_t* aY = sY + wrow * PD;

    for (int t = 0; t < n; ++t) {
        // ======================= layer 0 =======================
        {
            const bf16_t* aH = h0_rd + wrow * PH;
            // A fragments depend only on the K-tile: load once per layer.
            v16bf aF[6];
            #pragma unroll
            for (int kt = 0; kt < 6; ++kt)
                aF[kt] = (kt < 4) ? load_a_frag(aY, PD, ln, half, kt * 32)
                                  : load_a_frag(aH, PH, ln, half, (kt - 4) * 32);
            #pragma unroll
            for (int nh = 0; nh < 4; ++nh) {
                const int ni = nh * 16 + ln;
                v16bf Bi[2], Bf[2], Bg[2], Bo[2];
                auto ldB = [&](int kt, int b) {
                    const bf16_t* wb = (kt < 4) ? sWih0 : sWhh0;
                    const int str = (kt < 4) ? PD : PH;
                    const int k0  = (kt < 4) ? kt * 32 : (kt - 4) * 32;
                    Bi[b] = load_b_frag(wb, str, ni,        half, k0);
                    Bf[b] = load_b_frag(wb, str, 64  + ni,  half, k0);
                    Bg[b] = load_b_frag(wb, str, 128 + ni,  half, k0);
                    Bo[b] = load_b_frag(wb, str, 192 + ni,  half, k0);
                };
                v8f gi = {}, gf = {}, gg = {}, go = {};   // inline SRC2=0 start
                ldB(0, 0);
                #pragma unroll
                for (int kt = 0; kt < 6; ++kt) {          // pipelined K loop
                    const int cur = kt & 1;
                    if (kt + 1 < 6) ldB(kt + 1, cur ^ 1); // prefetch next tile
                    gi = WMMA_BF16(aF[kt], Bi[cur], gi);
                    gf = WMMA_BF16(aF[kt], Bf[cur], gf);
                    gg = WMMA_BF16(aF[kt], Bg[cur], gg);
                    go = WMMA_BF16(aF[kt], Bo[cur], go);
                }
                #pragma unroll
                for (int e = 0; e < 8; ++e) {             // cell update (+bias)
                    float i_ = fast_sigmoid(gi[e] + bi0[nh]);
                    float f_ = fast_sigmoid(gf[e] + bf0[nh]);
                    float g_ = fast_tanh  (gg[e] + bg0[nh]);
                    float o_ = fast_sigmoid(go[e] + bo0[nh]);
                    float c  = f_ * c0[nh][e] + i_ * g_;
                    c0[nh][e] = c;
                    float h  = o_ * fast_tanh(c);
                    const int m = e + 8 * half;           // D-layout row
                    h0_wr[(wrow + m) * PH + nh * 16 + ln] = (bf16_t)h;
                }
            }
        }
        // ======================= layer 1 =======================
        {
            const bf16_t* aHin = h0_wr + wrow * PH;       // new h0 (same wave)
            const bf16_t* aH   = h1_rd + wrow * PH;       // old h1
            v16bf aF[4];
            #pragma unroll
            for (int kt = 0; kt < 4; ++kt)
                aF[kt] = (kt < 2) ? load_a_frag(aHin, PH, ln, half, kt * 32)
                                  : load_a_frag(aH,   PH, ln, half, (kt - 2) * 32);
            #pragma unroll
            for (int nh = 0; nh < 4; ++nh) {
                const int ni = nh * 16 + ln;
                v16bf Bi[2], Bf[2], Bg[2], Bo[2];
                auto ldB = [&](int kt, int b) {
                    const bf16_t* wb = (kt < 2) ? sWih1 : sWhh1;
                    const int k0 = (kt & 1) * 32;
                    Bi[b] = load_b_frag(wb, PH, ni,        half, k0);
                    Bf[b] = load_b_frag(wb, PH, 64  + ni,  half, k0);
                    Bg[b] = load_b_frag(wb, PH, 128 + ni,  half, k0);
                    Bo[b] = load_b_frag(wb, PH, 192 + ni,  half, k0);
                };
                v8f gi = {}, gf = {}, gg = {}, go = {};
                ldB(0, 0);
                #pragma unroll
                for (int kt = 0; kt < 4; ++kt) {
                    const int cur = kt & 1;
                    if (kt + 1 < 4) ldB(kt + 1, cur ^ 1);
                    gi = WMMA_BF16(aF[kt], Bi[cur], gi);
                    gf = WMMA_BF16(aF[kt], Bf[cur], gf);
                    gg = WMMA_BF16(aF[kt], Bg[cur], gg);
                    go = WMMA_BF16(aF[kt], Bo[cur], go);
                }
                #pragma unroll
                for (int e = 0; e < 8; ++e) {
                    float i_ = fast_sigmoid(gi[e] + bi1[nh]);
                    float f_ = fast_sigmoid(gf[e] + bf1[nh]);
                    float g_ = fast_tanh  (gg[e] + bg1[nh]);
                    float o_ = fast_sigmoid(go[e] + bo1[nh]);
                    float c  = f_ * c1[nh][e] + i_ * g_;
                    c1[nh][e] = c;
                    float h  = o_ * fast_tanh(c);
                    const int m = e + 8 * half;
                    h1_wr[(wrow + m) * PH + nh * 16 + ln] = (bf16_t)h;
                }
            }
        }
        // ======================= fc projection =======================
        {
            const bf16_t* aH = h1_wr + wrow * PH;         // new h1 (same wave)
            float* ybase = y_stack + (size_t)t * plane + row0 * LSTM_D;
            v16bf aF[2];
            aF[0] = load_a_frag(aH, PH, ln, half, 0);
            aF[1] = load_a_frag(aH, PH, ln, half, 32);
            v16bf B0[2], B1[2];
            auto ldB = [&](int ny, int b) {
                B0[b] = load_b_frag(sWfc, PH, ny * 16 + ln, half, 0);
                B1[b] = load_b_frag(sWfc, PH, ny * 16 + ln, half, 32);
            };
            ldB(0, 0);
            #pragma unroll
            for (int ny = 0; ny < 8; ++ny) {              // pipelined over ny
                const int cur = ny & 1;
                if (ny + 1 < 8) ldB(ny + 1, cur ^ 1);
                v8f y = {};
                y = WMMA_BF16(aF[0], B0[cur], y);
                y = WMMA_BF16(aF[1], B1[cur], y);
                #pragma unroll
                for (int e = 0; e < 8; ++e) {
                    const int m = e + 8 * half;
                    const float v = y[e] + byc[ny];
                    const size_t gofs = (size_t)(wrow + m) * LSTM_D + ny * 16 + ln;
                    ybase[gofs] = v;                                    // y_stack[t]
                    sY[(wrow + m) * PD + ny * 16 + ln] = (bf16_t)v;     // feedback
                    if (t == n - 1) y_last[row0 * LSTM_D + gofs] = v;   // y_last
                }
            }
        }
        // ping-pong the recurrent h buffers
        bf16_t* tmp;
        tmp = h0_rd; h0_rd = h0_wr; h0_wr = tmp;
        tmp = h1_rd; h1_rd = h1_wr; h1_wr = tmp;
    }
}

extern "C" void kernel_launch(void* const* d_in, const int* in_sizes, int n_in,
                              void* d_out, int out_size, void* d_ws, size_t ws_size,
                              hipStream_t stream) {
    (void)n_in; (void)out_size; (void)d_ws; (void)ws_size;
    const float* x    = (const float*)d_in[0];
    const float* Wih0 = (const float*)d_in[1];
    const float* Whh0 = (const float*)d_in[2];
    const float* b0   = (const float*)d_in[3];
    const float* Wih1 = (const float*)d_in[4];
    const float* Whh1 = (const float*)d_in[5];
    const float* b1   = (const float*)d_in[6];
    const float* Wfc  = (const float*)d_in[7];
    const float* bfc  = (const float*)d_in[8];
    const int*   np   = (const int*)d_in[9];
    float* out = (float*)d_out;

    const int Btot = in_sizes[0] / LSTM_D;   // 32768
    const int grid = Btot / BT;              // 256 workgroups

    // Allow > 64KB dynamic LDS (CDNA5: 320KB per WGP). Idempotent per call.
    hipFuncSetAttribute((const void*)lstm2_wmma_kernel,
                        hipFuncAttributeMaxDynamicSharedMemorySize,
                        (int)SMEM_BYTES);

    lstm2_wmma_kernel<<<grid, NTHREADS, SMEM_BYTES, stream>>>(
        x, Wih0, Whh0, b0, Wih1, Whh1, b1, Wfc, bfc, np, out, Btot);
}
